// Resmambakan_53068615909924
// MI455X (gfx1250) — matrix operand order repaired and checked by reading, not compiled, not run on hardware
//
#include <hip/hip_runtime.h>
#include <hip/hip_bf16.h>
#include <math.h>

// ---------------------------------------------------------------------------
// Shapes (compile-time constants from the reference)
// ---------------------------------------------------------------------------
#define BATCH     4
#define DMODEL    128
#define DINNER    256
#define DSTATE    16
#define DCONV     4
#define DTRANK    8
#define HH        64
#define WW        64
#define NTOK      (HH*WW)          // 4096
#define MROWS     (BATCH*NTOK)     // 16384
#define NSPL      5                // GRID_SIZE + SPLINE_ORDER
#define KSPL      (DMODEL*NSPL)    // 640
#define XDBL_PAD  64               // 40 -> 64 (multiple of 64 for GEMM N)
#define DT_KPAD   32               // 8  -> 32

typedef __attribute__((ext_vector_type(16))) _Float16 v16h;
typedef __attribute__((ext_vector_type(8)))  float    v8f;

__device__ __forceinline__ float silu_f(float v) { return v / (1.f + __expf(-v)); }

#define WMMA_F16(a, b, c) \
    __builtin_amdgcn_wmma_f32_16x16x32_f16(false, (a), false, (b), (short)0, (c), false, false)

// ---------------------------------------------------------------------------
// WMMA f16 GEMM:  C[M x N] = A[M x K] * Bsw (+bias)(+addsrc)(act)
// - One wave computes a 16x64 strip: 4 accumulators share one A fragment.
// - 4 waves per block cover a 64-row band.  K % 32 == 0, N % 64 == 0,
//   M % 64 == 0.
// - B is pre-swizzled into fragment order so each lane reads one contiguous
//   32-byte run per tile (coalesced global_load_b128 pairs).
// - A fragment layout per CDNA5 ISA 7.12.2 (two contiguous 8-half runs).
// - Ping-pong double buffering over two disjoint register sets: no fragment
//   copies; WMMA groups consume the registers their loads targeted while the
//   other buffer's loads are in flight.
// ACT: 0 = none, 1 = softplus.
// ---------------------------------------------------------------------------
template<bool HAS_BIAS, bool HAS_ADD, int ACT>
__global__ __launch_bounds__(128)
void gemm_wmma_f16(const _Float16* __restrict__ A, const _Float16* __restrict__ Bsw,
                   const float* __restrict__ bias, const float* __restrict__ addsrc,
                   float* __restrict__ C, int M, int N, int K)
{
    const int lane = threadIdx.x & 31;
    const int wave = threadIdx.x >> 5;
    const int row0 = (blockIdx.y * 4 + wave) * 16;
    const int colt = blockIdx.x;            // 64-wide column tile
    const int col0 = colt * 64;
    const int nt16 = N >> 4;                // # of 16-wide col tiles
    const int KT   = K >> 5;                // # of 32-deep k tiles

    const int m  = lane & 15;
    const int ka = (lane < 16) ? 0 : 8;     // A: K sub-block per half-wave

    const _Float16* arow = A + (size_t)(row0 + m) * K;

    auto load_frags = [&](int kt, v16h& af, v16h& b0, v16h& b1, v16h& b2, v16h& b3) {
        const _Float16* ap = arow + kt * 32;
        #pragma unroll
        for (int i = 0; i < 8; ++i) {
            af[i]     = ap[ka + i];          // K = ka..ka+7      (b128)
            af[8 + i] = ap[16 + ka + i];     // K = 16+ka..+7     (b128)
        }
        const _Float16* bp = Bsw + ((size_t)(kt * nt16 + colt * 4) * 32 + lane) * 16;
        b0 = *(const v16h*)(bp);
        b1 = *(const v16h*)(bp + 32 * 16);
        b2 = *(const v16h*)(bp + 2 * 32 * 16);
        b3 = *(const v16h*)(bp + 3 * 32 * 16);
    };

    v8f acc0 = {}, acc1 = {}, acc2 = {}, acc3 = {};
    v16h aA, a0, a1, a2, a3;                // buffer A
    v16h aB, b0, b1, b2, b3;                // buffer B

    load_frags(0, aA, a0, a1, a2, a3);

    int kt = 0;
    for (; kt + 1 < KT; kt += 2) {
        load_frags(kt + 1, aB, b0, b1, b2, b3);
        if (kt + 2 < KT) __builtin_prefetch(arow + (kt + 2) * 32, 0, 3);
        acc0 = WMMA_F16(aA, a0, acc0);
        acc1 = WMMA_F16(aA, a1, acc1);
        acc2 = WMMA_F16(aA, a2, acc2);
        acc3 = WMMA_F16(aA, a3, acc3);
        if (kt + 2 < KT) load_frags(kt + 2, aA, a0, a1, a2, a3);
        acc0 = WMMA_F16(aB, b0, acc0);
        acc1 = WMMA_F16(aB, b1, acc1);
        acc2 = WMMA_F16(aB, b2, acc2);
        acc3 = WMMA_F16(aB, b3, acc3);
    }
    if (kt < KT) {                           // odd tail (covers KT == 1)
        acc0 = WMMA_F16(aA, a0, acc0);
        acc1 = WMMA_F16(aA, a1, acc1);
        acc2 = WMMA_F16(aA, a2, acc2);
        acc3 = WMMA_F16(aA, a3, acc3);
    }

    const int n  = lane & 15;
    const int mo = (lane < 16) ? 0 : 8;
    v8f accs[4] = {acc0, acc1, acc2, acc3};
    #pragma unroll
    for (int t = 0; t < 4; ++t) {
        const int col = col0 + t * 16 + n;
        #pragma unroll
        for (int r = 0; r < 8; ++r) {
            const int row = row0 + mo + r;
            float v = accs[t][r];
            if (HAS_BIAS) v += bias[col];
            if (HAS_ADD)  v += addsrc[(size_t)row * N + col];
            if (ACT == 1) v = (v > 20.f) ? v : log1pf(__expf(v));  // softplus
            C[(size_t)row * N + col] = v;
        }
    }
}

// ---------------------------------------------------------------------------
// Weight conversion: W[Nout x Kin] (row-major f32) -> fragment-swizzled f16
// B matrix, zero-padded.  Output index = ((kt*NT + nt)*32 + lane)*16 + i.
// ---------------------------------------------------------------------------
__global__ void cvt_weightT_sw(const float* __restrict__ W, _Float16* __restrict__ Bsw,
                               int Nout, int Kin, int Npad, int Kpad)
{
    int idx = blockIdx.x * blockDim.x + threadIdx.x;
    const int NT = Npad >> 4, KT = Kpad >> 5;
    if (idx >= KT * NT * 512) return;
    int i    = idx & 15;
    int lane = (idx >> 4) & 31;
    int tile = idx >> 9;
    int nt = tile % NT, kt = tile / NT;
    int nn = (nt << 4) + (lane & 15);
    int k  = (kt << 5) + ((lane < 16) ? 0 : 8) + (i & 7) + ((i & 8) ? 16 : 0);
    float v = (k < Kin && nn < Nout) ? W[(size_t)nn * Kin + k] : 0.f;
    Bsw[idx] = (_Float16)v;
}

// ---------------------------------------------------------------------------
// LayerNorm over C=128; one wave per token row; input NCHW, output f16 [r][c].
// ---------------------------------------------------------------------------
__global__ __launch_bounds__(32)
void ln_kernel(const float* __restrict__ x, const float* __restrict__ g,
               const float* __restrict__ bb, _Float16* __restrict__ tn)
{
    const int r = blockIdx.x;
    const int lane = threadIdx.x;
    const int b = r >> 12, nt = r & (NTOK - 1);
    const float* xp = x + ((size_t)b * DMODEL) * NTOK + nt;

    float v[4], s = 0.f, s2 = 0.f;
    #pragma unroll
    for (int i = 0; i < 4; ++i) {
        int c = lane + i * 32;
        float t = xp[(size_t)c * NTOK];
        v[i] = t; s += t; s2 += t * t;
    }
    #pragma unroll
    for (int msk = 16; msk >= 1; msk >>= 1) {
        s  += __shfl_xor(s,  msk, 32);
        s2 += __shfl_xor(s2, msk, 32);
    }
    float mean = s * (1.f / DMODEL);
    float var  = s2 * (1.f / DMODEL) - mean * mean;
    float inv  = rsqrtf(var + 1e-5f);
    #pragma unroll
    for (int i = 0; i < 4; ++i) {
        int c = lane + i * 32;
        tn[(size_t)r * DMODEL + c] = (_Float16)((v[i] - mean) * inv * g[c] + bb[c]);
    }
}

// ---------------------------------------------------------------------------
// Causal depthwise conv1d over L (kernel 4) on xm = xz[:, :256], + SiLU.
// ---------------------------------------------------------------------------
__global__ void conv1d_silu(const float* __restrict__ xz, const float* __restrict__ w,
                            const float* __restrict__ cb, float* __restrict__ xm,
                            _Float16* __restrict__ xmh)
{
    int idx = blockIdx.x * blockDim.x + threadIdx.x;
    if (idx >= MROWS * DINNER) return;
    int c = idx & (DINNER - 1);
    int r = idx >> 8;
    int b = r >> 12, l = r & (NTOK - 1);
    float acc = cb[c];
    #pragma unroll
    for (int j = 0; j < DCONV; ++j) {
        int ll = l - (DCONV - 1) + j;
        if (ll >= 0)
            acc += w[c * DCONV + j] * xz[((size_t)(b * NTOK + ll)) * (2 * DINNER) + c];
    }
    float sv = silu_f(acc);
    xm[idx]  = sv;
    xmh[idx] = (_Float16)sv;
}

// ---------------------------------------------------------------------------
// Pack dt (first 8 cols of x_dbl) into a zero-padded f16 M x 32 A matrix.
// ---------------------------------------------------------------------------
__global__ void pack_dt(const float* __restrict__ xdbl, _Float16* __restrict__ dth)
{
    int idx = blockIdx.x * blockDim.x + threadIdx.x;
    if (idx >= MROWS * DT_KPAD) return;
    int j = idx & (DT_KPAD - 1);
    int r = idx >> 5;
    dth[idx] = (_Float16)((j < DTRANK) ? xdbl[(size_t)r * XDBL_PAD + j] : 0.f);
}

// ---------------------------------------------------------------------------
// Selective scan: block per batch, thread per inner channel d, 16-state
// recurrence in registers; B/C vectors staged via LDS per step.
// Fuses + Dp*xm and * SiLU(z); writes f16 y for the out_proj GEMM.
// ---------------------------------------------------------------------------
__global__ __launch_bounds__(DINNER)
void scan_kernel(const float* __restrict__ dtf, const float* __restrict__ xdbl,
                 const float* __restrict__ xm,  const float* __restrict__ xz,
                 const float* __restrict__ A_log, const float* __restrict__ Dp,
                 _Float16* __restrict__ yh)
{
    const int b = blockIdx.x;
    const int d = threadIdx.x;
    __shared__ float sB[DSTATE], sC[DSTATE];

    float a[DSTATE], h[DSTATE];
    #pragma unroll
    for (int s = 0; s < DSTATE; ++s) {
        a[s] = -__expf(A_log[d * DSTATE + s]);
        h[s] = 0.f;
    }
    const float dp = Dp[d];

    for (int l = 0; l < NTOK; ++l) {
        const int r = (b << 12) | l;
        if (d < DSTATE)          sB[d]          = xdbl[(size_t)r * XDBL_PAD + DTRANK + d];
        else if (d < 2 * DSTATE) sC[d - DSTATE] = xdbl[(size_t)r * XDBL_PAD + DTRANK + DSTATE + (d - DSTATE)];
        __syncthreads();

        const float dtv = dtf[(size_t)r * DINNER + d];
        const float xv  = xm [(size_t)r * DINNER + d];
        float acc = 0.f;
        #pragma unroll
        for (int s = 0; s < DSTATE; ++s) {
            h[s] = __expf(dtv * a[s]) * h[s] + dtv * sB[s] * xv;
            acc += h[s] * sC[s];
        }
        const float zv = xz[(size_t)r * (2 * DINNER) + DINNER + d];
        const float y  = (acc + dp * xv) * silu_f(zv);
        yh[(size_t)r * DINNER + d] = (_Float16)y;
        __syncthreads();
    }
}

// ---------------------------------------------------------------------------
// tm = mamba_out + skip * t  (t read from original NCHW input); emits f16.
// ---------------------------------------------------------------------------
__global__ void residual_tm(const float* __restrict__ mout, const float* __restrict__ x,
                            const float* __restrict__ skip, _Float16* __restrict__ tmh)
{
    int idx = blockIdx.x * blockDim.x + threadIdx.x;
    if (idx >= MROWS * DMODEL) return;
    int c = idx & (DMODEL - 1);
    int r = idx >> 7;
    int b = r >> 12, nt = r & (NTOK - 1);
    float t = x[(((size_t)b * DMODEL + c) << 12) + nt];
    tmh[idx] = (_Float16)(mout[idx] + skip[0] * t);
}

// ---------------------------------------------------------------------------
// KAN pre: depthwise 3x3 conv + BN + ReLU on out1; emits t2 (f32) and
// SiLU(t2) (f16) for the base GEMM.
// ---------------------------------------------------------------------------
__global__ void kan_pre(const float* __restrict__ out1, const float* __restrict__ w,
                        const float* __restrict__ wb, const float* __restrict__ g,
                        const float* __restrict__ bb, const float* __restrict__ mean,
                        const float* __restrict__ var, float* __restrict__ t2,
                        _Float16* __restrict__ sh)
{
    int idx = blockIdx.x * blockDim.x + threadIdx.x;
    if (idx >= BATCH * DMODEL * NTOK) return;
    int p = idx & (NTOK - 1);
    int c = (idx >> 12) & (DMODEL - 1);
    int b = idx >> 19;
    int hy = p >> 6, wx = p & 63;

    float acc = wb[c];
    #pragma unroll
    for (int dh = 0; dh < 3; ++dh)
        #pragma unroll
        for (int dw = 0; dw < 3; ++dw) {
            int yy = hy + dh - 1, xx = wx + dw - 1;
            if (yy >= 0 && yy < HH && xx >= 0 && xx < WW)
                acc += w[c * 9 + dh * 3 + dw] *
                       out1[((size_t)(b * NTOK + (yy << 6) + xx)) * DMODEL + c];
        }
    float sc = g[c] * rsqrtf(var[c] + 1e-5f);
    float v  = acc * sc + (bb[c] - mean[c] * sc);
    v = v > 0.f ? v : 0.f;
    size_t o = ((size_t)(b * NTOK) + p) * DMODEL + c;
    t2[o] = v;
    sh[o] = (_Float16)silu_f(v);
}

// ---------------------------------------------------------------------------
// Order-2 Cox–de Boor B-spline bases: 5 per (row, channel), packed f16
// into the M x 640 A-matrix (column index = c*5 + j).
// ---------------------------------------------------------------------------
__global__ void spline_kernel(const float* __restrict__ t2, const float* __restrict__ grid,
                              _Float16* __restrict__ bh)
{
    int idx = blockIdx.x * blockDim.x + threadIdx.x;
    if (idx >= MROWS * DMODEL) return;
    int c = idx & (DMODEL - 1);
    int r = idx >> 7;
    float x = t2[idx];
    const float* g = grid + c * 8;

    float b0[7], b1[6], b2[5];
    #pragma unroll
    for (int j = 0; j < 7; ++j) b0[j] = (x >= g[j] && x < g[j + 1]) ? 1.f : 0.f;
    #pragma unroll
    for (int j = 0; j < 6; ++j)
        b1[j] = (x - g[j]) / (g[j + 1] - g[j]) * b0[j] +
                (g[j + 2] - x) / (g[j + 2] - g[j + 1]) * b0[j + 1];
    #pragma unroll
    for (int j = 0; j < 5; ++j)
        b2[j] = (x - g[j]) / (g[j + 2] - g[j]) * b1[j] +
                (g[j + 3] - x) / (g[j + 3] - g[j + 1]) * b1[j + 1];

    size_t base = (size_t)r * KSPL + c * NSPL;
    #pragma unroll
    for (int j = 0; j < 5; ++j) bh[base + j] = (_Float16)b2[j];
}

// ---------------------------------------------------------------------------
// Final: out(NCHW) = out1 + (base + spline), token-major -> NCHW.
// ---------------------------------------------------------------------------
__global__ void final_kernel(const float* __restrict__ out1, const float* __restrict__ ktot,
                             float* __restrict__ out)
{
    int idx = blockIdx.x * blockDim.x + threadIdx.x;
    if (idx >= BATCH * DMODEL * NTOK) return;
    int p = idx & (NTOK - 1);
    int c = (idx >> 12) & (DMODEL - 1);
    int b = idx >> 19;
    size_t r = (size_t)b * NTOK + p;
    out[idx] = out1[r * DMODEL + c] + ktot[r * DMODEL + c];
}

// ---------------------------------------------------------------------------
// Host launcher
// ---------------------------------------------------------------------------
extern "C" void kernel_launch(void* const* d_in, const int* in_sizes, int n_in,
                              void* d_out, int out_size, void* d_ws, size_t ws_size,
                              hipStream_t stream)
{
    (void)in_sizes; (void)n_in; (void)out_size; (void)ws_size;

    const float* x         = (const float*)d_in[0];
    const float* ln_g      = (const float*)d_in[1];
    const float* ln_b      = (const float*)d_in[2];
    const float* in_proj_w = (const float*)d_in[3];
    const float* conv_w    = (const float*)d_in[4];
    const float* conv_b    = (const float*)d_in[5];
    const float* x_proj_w  = (const float*)d_in[6];
    const float* dt_proj_w = (const float*)d_in[7];
    const float* dt_proj_b = (const float*)d_in[8];
    const float* A_log     = (const float*)d_in[9];
    const float* Dp        = (const float*)d_in[10];
    const float* out_proj_w= (const float*)d_in[11];
    const float* skip      = (const float*)d_in[12];
    const float* proj_w    = (const float*)d_in[13];
    const float* proj_b    = (const float*)d_in[14];
    const float* dw_w      = (const float*)d_in[15];
    const float* dw_b      = (const float*)d_in[16];
    const float* bn_g      = (const float*)d_in[17];
    const float* bn_b      = (const float*)d_in[18];
    const float* bn_mean   = (const float*)d_in[19];
    const float* bn_var    = (const float*)d_in[20];
    const float* grid      = (const float*)d_in[21];
    const float* base_w    = (const float*)d_in[22];
    const float* spline_w  = (const float*)d_in[23];
    float* out = (float*)d_out;

    char* ws = (char*)d_ws;
    size_t off = 0;
    auto alloc = [&](size_t bytes) -> void* {
        off = (off + 255) & ~(size_t)255;
        void* p = ws + off;
        off += bytes;
        return p;
    };

    _Float16* tn_h   = (_Float16*)alloc((size_t)MROWS * DMODEL * 2);
    _Float16* B1     = (_Float16*)alloc((size_t)DMODEL * (2*DINNER) * 2);   // 128x512
    _Float16* B2     = (_Float16*)alloc((size_t)DINNER * XDBL_PAD * 2);     // 256x64
    _Float16* B3     = (_Float16*)alloc((size_t)DT_KPAD * DINNER * 2);      // 32x256
    _Float16* B4     = (_Float16*)alloc((size_t)DINNER * DMODEL * 2);       // 256x128
    _Float16* B5     = (_Float16*)alloc((size_t)DMODEL * DMODEL * 2);       // 128x128
    _Float16* B6     = (_Float16*)alloc((size_t)DMODEL * DMODEL * 2);       // 128x128
    _Float16* B7     = (_Float16*)alloc((size_t)KSPL   * DMODEL * 2);       // 640x128
    float*    xz     = (float*)   alloc((size_t)MROWS * 2*DINNER * 4);
    float*    xm_f   = (float*)   alloc((size_t)MROWS * DINNER * 4);
    _Float16* xm_h   = (_Float16*)alloc((size_t)MROWS * DINNER * 2);
    float*    xdbl   = (float*)   alloc((size_t)MROWS * XDBL_PAD * 4);
    _Float16* dt_h   = (_Float16*)alloc((size_t)MROWS * DT_KPAD * 2);
    float*    dtf    = (float*)   alloc((size_t)MROWS * DINNER * 4);
    _Float16* y_h    = (_Float16*)alloc((size_t)MROWS * DINNER * 2);
    float*    mout   = (float*)   alloc((size_t)MROWS * DMODEL * 4);
    _Float16* tm_h   = (_Float16*)alloc((size_t)MROWS * DMODEL * 2);
    float*    out1   = (float*)   alloc((size_t)MROWS * DMODEL * 4);
    float*    t2     = (float*)   alloc((size_t)MROWS * DMODEL * 4);
    _Float16* silu_h = (_Float16*)alloc((size_t)MROWS * DMODEL * 2);
    _Float16* bas_h  = (_Float16*)alloc((size_t)MROWS * KSPL * 2);
    float*    kbase  = (float*)   alloc((size_t)MROWS * DMODEL * 4);
    float*    ktot   = (float*)   alloc((size_t)MROWS * DMODEL * 4);

    auto cvt = [&](const float* W, _Float16* Bm, int Nout, int Kin, int Npad, int Kpad) {
        int total = Npad * Kpad;
        cvt_weightT_sw<<<(total + 255) / 256, 256, 0, stream>>>(W, Bm, Nout, Kin, Npad, Kpad);
    };

    // Weight transposes into fragment-swizzled f16 layout
    cvt(in_proj_w,  B1, 2*DINNER, DMODEL,  2*DINNER, DMODEL);
    cvt(x_proj_w,   B2, DTRANK + 2*DSTATE, DINNER, XDBL_PAD, DINNER);
    cvt(dt_proj_w,  B3, DINNER,  DTRANK,  DINNER,   DT_KPAD);
    cvt(out_proj_w, B4, DMODEL,  DINNER,  DMODEL,   DINNER);
    cvt(proj_w,     B5, DMODEL,  DMODEL,  DMODEL,   DMODEL);
    cvt(base_w,     B6, DMODEL,  DMODEL,  DMODEL,   DMODEL);
    cvt(spline_w,   B7, DMODEL,  KSPL,    DMODEL,   KSPL);

    // 1. LayerNorm -> tn (f16)
    ln_kernel<<<MROWS, 32, 0, stream>>>(x, ln_g, ln_b, tn_h);

    // 2. xz = tn @ in_proj^T   (16384 x 512, K=128)
    gemm_wmma_f16<false,false,0><<<dim3((2*DINNER)/64, MROWS/64), 128, 0, stream>>>(
        tn_h, B1, nullptr, nullptr, xz, MROWS, 2*DINNER, DMODEL);

    // 3. causal dwconv1d + SiLU -> xm
    conv1d_silu<<<(MROWS*DINNER + 255) / 256, 256, 0, stream>>>(xz, conv_w, conv_b, xm_f, xm_h);

    // 4. x_dbl = xm @ x_proj^T  (16384 x 64, K=256)
    gemm_wmma_f16<false,false,0><<<dim3(XDBL_PAD/64, MROWS/64), 128, 0, stream>>>(
        xm_h, B2, nullptr, nullptr, xdbl, MROWS, XDBL_PAD, DINNER);

    // 5. pack dt rank-8 slice into padded f16 A
    pack_dt<<<(MROWS*DT_KPAD + 255) / 256, 256, 0, stream>>>(xdbl, dt_h);

    // 6. dt = softplus(dt @ dt_proj^T + b)  (16384 x 256, K=32)
    gemm_wmma_f16<true,false,1><<<dim3(DINNER/64, MROWS/64), 128, 0, stream>>>(
        dt_h, B3, dt_proj_b, nullptr, dtf, MROWS, DINNER, DT_KPAD);

    // 7. selective scan (+ D*x, * SiLU(z)) -> y (f16)
    scan_kernel<<<BATCH, DINNER, 0, stream>>>(dtf, xdbl, xm_f, xz, A_log, Dp, y_h);

    // 8. mamba out = y @ out_proj^T  (16384 x 128, K=256)
    gemm_wmma_f16<false,false,0><<<dim3(DMODEL/64, MROWS/64), 128, 0, stream>>>(
        y_h, B4, nullptr, nullptr, mout, MROWS, DMODEL, DINNER);

    // 9. tm = mout + skip * t  (f16)
    residual_tm<<<(MROWS*DMODEL + 255) / 256, 256, 0, stream>>>(mout, x, skip, tm_h);

    // 10. out1 = tm @ proj^T + proj_b  (16384 x 128, K=128)
    gemm_wmma_f16<true,false,0><<<dim3(DMODEL/64, MROWS/64), 128, 0, stream>>>(
        tm_h, B5, proj_b, nullptr, out1, MROWS, DMODEL, DMODEL);

    // 11. KAN pre: dwconv3x3 + BN + ReLU -> t2, SiLU(t2) f16
    kan_pre<<<(BATCH*DMODEL*NTOK + 255) / 256, 256, 0, stream>>>(
        out1, dw_w, dw_b, bn_g, bn_b, bn_mean, bn_var, t2, silu_h);

    // 12. spline bases (f16, 16384 x 640)
    spline_kernel<<<(MROWS*DMODEL + 255) / 256, 256, 0, stream>>>(t2, grid, bas_h);

    // 13. base = SiLU(t2) @ base_w^T  (16384 x 128, K=128)
    gemm_wmma_f16<false,false,0><<<dim3(DMODEL/64, MROWS/64), 128, 0, stream>>>(
        silu_h, B6, nullptr, nullptr, kbase, MROWS, DMODEL, DMODEL);

    // 14. ktot = bases @ spline_w^T + base  (16384 x 128, K=640)
    gemm_wmma_f16<false,true,0><<<dim3(DMODEL/64, MROWS/64), 128, 0, stream>>>(
        bas_h, B7, nullptr, kbase, ktot, MROWS, DMODEL, KSPL);

    // 15. out = out1 + ktot  (token-major -> NCHW)
    final_kernel<<<(BATCH*DMODEL*NTOK + 255) / 256, 256, 0, stream>>>(out1, ktot, out);
}
